// LargeGraphDualEncoder_41334765256945
// MI455X (gfx1250) — matrix operand order repaired and checked by reading, not compiled, not run on hardware
//
#include <hip/hip_runtime.h>
#include <hip/hip_bf16.h>
#include <math.h>

// ---------------- problem constants (from reference) ----------------
#define LNUM   6
#define HEADS  4
#define CH     128
#define HID    512
#define INDIM  768
#define NNODE  20000
#define NEDGE  320000
#define NPAIR  100000

typedef __attribute__((ext_vector_type(16))) __bf16 v16bf;
typedef __attribute__((ext_vector_type(8)))  __bf16 v8bf;
typedef __attribute__((ext_vector_type(8)))  float  v8f;

// ---- async global->LDS staging (gfx1250 ASYNCcnt path), guarded ----
#if defined(__AMDGCN__) && defined(__gfx1250__) && \
    __has_builtin(__builtin_amdgcn_global_load_async_to_lds_b128)
#define HAVE_ASYNC_LDS 1
#else
#define HAVE_ASYNC_LDS 0
#endif

#if HAVE_ASYNC_LDS
// builtin prototype: void(v4i32 AS1*, v4i32 AS3*, imm int offset, imm int cpol)
typedef int v4i_vs __attribute__((vector_size(16)));
typedef __attribute__((address_space(1))) v4i_vs* gv4p_t;
typedef __attribute__((address_space(3))) v4i_vs* lv4p_t;
__device__ __forceinline__ void async_copy16(const float* g, float* l) {
    __builtin_amdgcn_global_load_async_to_lds_b128((gv4p_t)g, (lv4p_t)l, 0, 0);
}
__device__ __forceinline__ void wait_async0() {
#if __has_builtin(__builtin_amdgcn_s_wait_asynccnt)
    __builtin_amdgcn_s_wait_asynccnt(0);
#else
    asm volatile("s_wait_asynccnt 0x0" ::: "memory");
#endif
}
#endif

__device__ __forceinline__ float gelu_f(float x) {
    return 0.5f * x * (1.0f + erff(x * 0.70710678118654752f));
}

__device__ __forceinline__ void atomic_max_float(float* addr, float val) {
    float old = *addr;
    while (old < val) {
        float assumed = old;
        old = __uint_as_float(atomicCAS((unsigned int*)addr,
                                        __float_as_uint(assumed),
                                        __float_as_uint(val)));
        if (old == assumed) break;
    }
}

// Fragment builders per CDNA5 ISA VGPR layouts (wave32).
__device__ __forceinline__ v16bf load_afrag(const __bf16* As, int row, int half) {
    v8bf lo = *(const v8bf*)&As[row * 32 + half * 8];
    v8bf hi = *(const v8bf*)&As[row * 32 + 16 + half * 8];
    return __builtin_shufflevector(lo, hi, 0,1,2,3,4,5,6,7,8,9,10,11,12,13,14,15);
}
__device__ __forceinline__ v16bf load_bfrag(const __bf16* BsT, int col, int half) {
    v8bf lo = *(const v8bf*)&BsT[col * 32 + half * 8];
    v8bf hi = *(const v8bf*)&BsT[col * 32 + 16 + half * 8];
    return __builtin_shufflevector(lo, hi, 0,1,2,3,4,5,6,7,8,9,10,11,12,13,14,15);
}

// ---------------- generic GEMM: C = act(A[MxK] @ B[KxN] + bias) ----------------
// Block tile 128x64, 8 waves; each wave owns a 32x32 register tile = four
// 16x16 WMMA tiles (2 A-frags x 2 B-frags -> 4 v_wmma per K-step).
// B tile is staged through the async global->LDS datapath when available.
__global__ __launch_bounds__(256)
void gemm_bias_act_wmma(const float* __restrict__ A, const float* __restrict__ B,
                        const float* __restrict__ bias, float* __restrict__ C,
                        int M, int N, int K, int act)
{
    __shared__ __align__(16) __bf16 As[128 * 32];   // [row][k]
    __shared__ __align__(16) __bf16 BsT[64 * 32];   // [n][k] (K-transposed)
#if HAVE_ASYNC_LDS
    __shared__ __align__(16) float Bstage[32 * 64]; // raw fp32 B tile (async dest)
#endif

    const int tid  = threadIdx.x;
    const int lane = tid & 31;
    const int wid  = tid >> 5;
    const int wm   = wid & 3;      // 0..3 -> 32-row group
    const int wn   = wid >> 2;     // 0..1 -> 32-col group
    const int half = lane >> 4;
    const int l16  = lane & 15;

    const int m_blk = blockIdx.x * 128;
    const int n_blk = blockIdx.y * 64;

    const int arow = tid >> 1;            // 0..127
    const int acol = (tid & 1) * 16;      // 0 or 16
    const int brow = tid >> 3;            // 0..31 (k)
    const int bcol = (tid & 7) * 8;       // 0..56 (n)

    // Clamp out-of-range rows instead of branching: garbage rows are never stored.
    int a_gr = m_blk + arow;
    if (a_gr > M - 1) a_gr = M - 1;
    const float* ap = A + (long)a_gr * K + acol;
    const float* bp = B + (long)brow * N + n_blk + bcol;

    v8f acc00 = {0.f,0.f,0.f,0.f,0.f,0.f,0.f,0.f};
    v8f acc01 = acc00, acc10 = acc00, acc11 = acc00;

    for (int k0 = 0; k0 < K; k0 += 32, ap += 32, bp += 32L * N) {
        // ---- stage A tile (128x32 fp32 -> bf16), branch-free ----
        {
            float4 f0 = *(const float4*)(ap + 0);
            float4 f1 = *(const float4*)(ap + 4);
            float4 f2 = *(const float4*)(ap + 8);
            float4 f3 = *(const float4*)(ap + 12);
            __builtin_prefetch(ap + 32, 0, 3);
            __bf16* as = &As[arow * 32 + acol];
            as[0]  = (__bf16)f0.x; as[1]  = (__bf16)f0.y; as[2]  = (__bf16)f0.z; as[3]  = (__bf16)f0.w;
            as[4]  = (__bf16)f1.x; as[5]  = (__bf16)f1.y; as[6]  = (__bf16)f1.z; as[7]  = (__bf16)f1.w;
            as[8]  = (__bf16)f2.x; as[9]  = (__bf16)f2.y; as[10] = (__bf16)f2.z; as[11] = (__bf16)f2.w;
            as[12] = (__bf16)f3.x; as[13] = (__bf16)f3.y; as[14] = (__bf16)f3.z; as[15] = (__bf16)f3.w;
        }
        // ---- stage B tile (32x64), async global->LDS, then K-transpose ----
        {
            float4 g0, g1;
#if HAVE_ASYNC_LDS
            float* st = &Bstage[brow * 64 + bcol];
            async_copy16(bp,     st);
            async_copy16(bp + 4, st + 4);
            __builtin_prefetch(bp + 32L * N, 0, 3);
            wait_async0();                       // our own slice has landed
            g0 = *(const float4*)st;
            g1 = *(const float4*)(st + 4);
#else
            g0 = *(const float4*)(bp + 0);
            g1 = *(const float4*)(bp + 4);
            __builtin_prefetch(bp + 32L * N, 0, 3);
#endif
            BsT[(bcol + 0) * 32 + brow] = (__bf16)g0.x;
            BsT[(bcol + 1) * 32 + brow] = (__bf16)g0.y;
            BsT[(bcol + 2) * 32 + brow] = (__bf16)g0.z;
            BsT[(bcol + 3) * 32 + brow] = (__bf16)g0.w;
            BsT[(bcol + 4) * 32 + brow] = (__bf16)g1.x;
            BsT[(bcol + 5) * 32 + brow] = (__bf16)g1.y;
            BsT[(bcol + 6) * 32 + brow] = (__bf16)g1.z;
            BsT[(bcol + 7) * 32 + brow] = (__bf16)g1.w;
        }
        __syncthreads();

        const int row0 = wm * 32 + l16;
        const int col0 = wn * 32 + l16;
        v16bf a0 = load_afrag(As, row0,      half);
        v16bf a1 = load_afrag(As, row0 + 16, half);
        v16bf b0 = load_bfrag(BsT, col0,      half);
        v16bf b1 = load_bfrag(BsT, col0 + 16, half);

        acc00 = __builtin_amdgcn_wmma_f32_16x16x32_bf16(false, a0, false, b0, (short)0, acc00, false, false);
        acc01 = __builtin_amdgcn_wmma_f32_16x16x32_bf16(false, a0, false, b1, (short)0, acc01, false, false);
        acc10 = __builtin_amdgcn_wmma_f32_16x16x32_bf16(false, a1, false, b0, (short)0, acc10, false, false);
        acc11 = __builtin_amdgcn_wmma_f32_16x16x32_bf16(false, a1, false, b1, (short)0, acc11, false, false);
        __syncthreads();
    }

    // ---- epilogue: bias + activation, store per C-layout ----
    const int gcol0 = n_blk + wn * 32 + l16;
    const int gcol1 = gcol0 + 16;
    const float bv0 = bias ? bias[gcol0] : 0.f;
    const float bv1 = bias ? bias[gcol1] : 0.f;
    #pragma unroll
    for (int r = 0; r < 8; ++r) {
        const int grow0 = m_blk + wm * 32 + r + half * 8;
        const int grow1 = grow0 + 16;
        if (grow0 < M) {
            float v0 = acc00[r] + bv0;
            float v1 = acc01[r] + bv1;
            if (act == 1) { v0 = gelu_f(v0); v1 = gelu_f(v1); }
            C[(long)grow0 * N + gcol0] = v0;
            C[(long)grow0 * N + gcol1] = v1;
        }
        if (grow1 < M) {
            float v0 = acc10[r] + bv0;
            float v1 = acc11[r] + bv1;
            if (act == 1) { v0 = gelu_f(v0); v1 = gelu_f(v1); }
            C[(long)grow1 * N + gcol0] = v0;
            C[(long)grow1 * N + gcol1] = v1;
        }
    }
}

// ---------------- fill helper ----------------
__global__ void fill_kernel(float* __restrict__ p, long n, float val) {
    long i = (long)blockIdx.x * blockDim.x + threadIdx.x;
    if (i < n) p[i] = val;
}

// ---------------- edge attention: logits + running max ----------------
__global__ __launch_bounds__(256)
void edge_logits_kernel(const int* __restrict__ ei, const float* __restrict__ q,
                        const float* __restrict__ k, float* __restrict__ logits,
                        float* __restrict__ mx)
{
    long t = (long)blockIdx.x * 256 + threadIdx.x;
    if (t >= (long)NEDGE * HEADS) return;
    int e  = (int)(t >> 2);
    int hd = (int)(t & 3);
    int src = ei[e], dst = ei[NEDGE + e];
    const float4* qp = (const float4*)(q + (long)dst * HID + hd * CH);
    const float4* kp = (const float4*)(k + (long)src * HID + hd * CH);
    float s = 0.f;
    #pragma unroll 4
    for (int u = 0; u < CH / 4; ++u) {
        float4 a = qp[u], b = kp[u];
        s += a.x * b.x + a.y * b.y + a.z * b.z + a.w * b.w;
    }
    s *= 0.08838834764831845f;  // 1/sqrt(128)
    logits[(long)e * HEADS + hd] = s;
    atomic_max_float(&mx[(long)dst * HEADS + hd], s);
}

// ---------------- edge attention: exp + denominator ----------------
__global__ __launch_bounds__(256)
void edge_exp_kernel(const int* __restrict__ ei, float* __restrict__ logits,
                     const float* __restrict__ mx, float* __restrict__ den)
{
    long t = (long)blockIdx.x * 256 + threadIdx.x;
    if (t >= (long)NEDGE * HEADS) return;
    int e  = (int)(t >> 2);
    int hd = (int)(t & 3);
    int dst = ei[NEDGE + e];
    float ex = expf(logits[(long)e * HEADS + hd] - mx[(long)dst * HEADS + hd]);
    logits[(long)e * HEADS + hd] = ex;
    atomicAdd(&den[(long)dst * HEADS + hd], ex);
}

// ---------------- edge attention: weighted message scatter ----------------
__global__ __launch_bounds__(256)
void edge_msg_kernel(const int* __restrict__ ei, const float* __restrict__ logits,
                     const float* __restrict__ den, const float* __restrict__ v,
                     float* __restrict__ agg)
{
    long t = (long)blockIdx.x * 256 + threadIdx.x;
    if (t >= (long)NEDGE * (HID / 4)) return;
    int e = (int)(t >> 7);
    int r = (int)(t & 127);
    int c = r * 4;
    int hd = c >> 7;
    int src = ei[e], dst = ei[NEDGE + e];
    float alpha = logits[(long)e * HEADS + hd] / den[(long)dst * HEADS + hd];
    float4 vv = *(const float4*)(v + (long)src * HID + c);
    float* o = agg + (long)dst * HID + c;
    atomicAdd(o + 0, alpha * vv.x);
    atomicAdd(o + 1, alpha * vv.y);
    atomicAdd(o + 2, alpha * vv.z);
    atomicAdd(o + 3, alpha * vv.w);
}

// ---------------- residual + GELU + LayerNorm (block per node) ----------------
__global__ __launch_bounds__(256)
void resid_ln_kernel(const float* __restrict__ pre, float* __restrict__ h,
                     const float* __restrict__ g, const float* __restrict__ b)
{
    const int n = blockIdx.x;
    const int t = threadIdx.x;
    __shared__ float red[256];
    const long base = (long)n * HID;
    float t0 = h[base + t]       + gelu_f(pre[base + t]);
    float t1 = h[base + t + 256] + gelu_f(pre[base + t + 256]);
    red[t] = t0 + t1;
    __syncthreads();
    for (int o = 128; o > 0; o >>= 1) {
        if (t < o) red[t] += red[t + o];
        __syncthreads();
    }
    const float mean = red[0] * (1.f / HID);
    __syncthreads();
    const float d0 = t0 - mean, d1 = t1 - mean;
    red[t] = d0 * d0 + d1 * d1;
    __syncthreads();
    for (int o = 128; o > 0; o >>= 1) {
        if (t < o) red[t] += red[t + o];
        __syncthreads();
    }
    const float inv = rsqrtf(red[0] * (1.f / HID) + 1e-5f);
    h[base + t]       = d0 * inv * g[t]       + b[t];
    h[base + t + 256] = d1 * inv * g[t + 256] + b[t + 256];
}

// ---------------- bilinear pair scores: s = dot(T[i], z[j]) ----------------
__global__ __launch_bounds__(256)
void pair_score_kernel(const int* __restrict__ edges,
                       const float* __restrict__ Tsyn, const float* __restrict__ zsyn,
                       const float* __restrict__ Tant, const float* __restrict__ zant,
                       float* __restrict__ ssyn, float* __restrict__ sant)
{
    int p = blockIdx.x * 256 + threadIdx.x;
    if (p >= NPAIR) return;
    int i = edges[p], j = edges[NPAIR + p];
    const float4* ti = (const float4*)(Tsyn + (long)i * HID);
    const float4* zj = (const float4*)(zsyn + (long)j * HID);
    float s = 0.f;
    for (int u = 0; u < HID / 4; ++u) {
        float4 a = ti[u], b = zj[u];
        s += a.x * b.x + a.y * b.y + a.z * b.z + a.w * b.w;
    }
    ssyn[p] = s;
    const float4* ta = (const float4*)(Tant + (long)i * HID);
    const float4* za = (const float4*)(zant + (long)j * HID);
    s = 0.f;
    for (int u = 0; u < HID / 4; ++u) {
        float4 a = ta[u], b = za[u];
        s += a.x * b.x + a.y * b.y + a.z * b.z + a.w * b.w;
    }
    sant[p] = s;
}

// ---------------- fused gate MLP ----------------
// hidden = gelu(pairfeat @ gW1 + gb1); gacc[p] += sum_n hidden * gW2
// Pair features (|zi-zj|, zi*zj for syn/ant) built on the fly into LDS.
__global__ __launch_bounds__(256)
void gate_wmma_kernel(const int* __restrict__ edges, const float* __restrict__ zs,
                      const float* __restrict__ za, const float* __restrict__ gW1,
                      const float* __restrict__ gb1, const float* __restrict__ gW2,
                      float* __restrict__ gacc)
{
    __shared__ __align__(16) __bf16 As[128 * 32];
    __shared__ __align__(16) __bf16 BsT[64 * 32];
#if HAVE_ASYNC_LDS
    __shared__ __align__(16) float Bstage[32 * 64];
#endif
    __shared__ float red[128 * 64];
    __shared__ int ii[128], jj[128];

    const int tid  = threadIdx.x;
    const int lane = tid & 31;
    const int wid  = tid >> 5;
    const int wm   = wid & 3;
    const int wn   = wid >> 2;
    const int half = lane >> 4;
    const int l16  = lane & 15;

    const int p_blk = blockIdx.x * 128;
    const int n_blk = blockIdx.y * 64;

    if (tid < 128) {
        int p = p_blk + tid;
        if (p > NPAIR - 1) p = NPAIR - 1;   // clamp; guarded at accumulate
        ii[tid] = edges[p];
        jj[tid] = edges[NPAIR + p];
    }
    __syncthreads();

    const int arow = tid >> 1;            // 0..127
    const int acol = (tid & 1) * 16;      // 0 or 16
    const int brow = tid >> 3;            // 0..31
    const int bcol = (tid & 7) * 8;       // 0..56

    const float* bp = gW1 + (long)brow * HID + n_blk + bcol;

    v8f acc00 = {0.f,0.f,0.f,0.f,0.f,0.f,0.f,0.f};
    v8f acc01 = acc00, acc10 = acc00, acc11 = acc00;

    for (int k0 = 0; k0 < 4 * HID; k0 += 32, bp += 32L * HID) {
        // ---- A tile: 16 consecutive pair-features of one pair row ----
        {
            const int f   = k0 + acol;        // 0..2047, multiple of 16
            const int seg = f >> 9;           // 0..3 (constant across 16)
            const int c   = f & (HID - 1);
            const float* zb = (seg < 2) ? zs : za;
            const float* zi = zb + (long)ii[arow] * HID + c;
            const float* zj = zb + (long)jj[arow] * HID + c;
            __bf16* as = &As[arow * 32 + acol];
            if (seg & 1) {
                #pragma unroll
                for (int u = 0; u < 16; ++u) as[u] = (__bf16)(zi[u] * zj[u]);
            } else {
                #pragma unroll
                for (int u = 0; u < 16; ++u) as[u] = (__bf16)fabsf(zi[u] - zj[u]);
            }
        }
        // ---- B tile of gW1 [2048 x 512], async stage + K-transpose ----
        {
            float4 g0, g1;
#if HAVE_ASYNC_LDS
            float* st = &Bstage[brow * 64 + bcol];
            async_copy16(bp,     st);
            async_copy16(bp + 4, st + 4);
            __builtin_prefetch(bp + 32L * HID, 0, 3);
            wait_async0();
            g0 = *(const float4*)st;
            g1 = *(const float4*)(st + 4);
#else
            g0 = *(const float4*)(bp + 0);
            g1 = *(const float4*)(bp + 4);
            __builtin_prefetch(bp + 32L * HID, 0, 3);
#endif
            BsT[(bcol + 0) * 32 + brow] = (__bf16)g0.x;
            BsT[(bcol + 1) * 32 + brow] = (__bf16)g0.y;
            BsT[(bcol + 2) * 32 + brow] = (__bf16)g0.z;
            BsT[(bcol + 3) * 32 + brow] = (__bf16)g0.w;
            BsT[(bcol + 4) * 32 + brow] = (__bf16)g1.x;
            BsT[(bcol + 5) * 32 + brow] = (__bf16)g1.y;
            BsT[(bcol + 6) * 32 + brow] = (__bf16)g1.z;
            BsT[(bcol + 7) * 32 + brow] = (__bf16)g1.w;
        }
        __syncthreads();

        const int row0 = wm * 32 + l16;
        const int col0 = wn * 32 + l16;
        v16bf a0 = load_afrag(As, row0,      half);
        v16bf a1 = load_afrag(As, row0 + 16, half);
        v16bf b0 = load_bfrag(BsT, col0,      half);
        v16bf b1 = load_bfrag(BsT, col0 + 16, half);

        acc00 = __builtin_amdgcn_wmma_f32_16x16x32_bf16(false, a0, false, b0, (short)0, acc00, false, false);
        acc01 = __builtin_amdgcn_wmma_f32_16x16x32_bf16(false, a0, false, b1, (short)0, acc01, false, false);
        acc10 = __builtin_amdgcn_wmma_f32_16x16x32_bf16(false, a1, false, b0, (short)0, acc10, false, false);
        acc11 = __builtin_amdgcn_wmma_f32_16x16x32_bf16(false, a1, false, b1, (short)0, acc11, false, false);
        __syncthreads();
    }

    // ---- epilogue: gelu(+gb1) * gW2 into LDS, row-reduce, atomicAdd ----
    const int gcol0 = n_blk + wn * 32 + l16;
    const int gcol1 = gcol0 + 16;
    const float b10 = gb1[gcol0], w20 = gW2[gcol0];
    const float b11 = gb1[gcol1], w21 = gW2[gcol1];
    const int lc0 = wn * 32 + l16;
    #pragma unroll
    for (int r = 0; r < 8; ++r) {
        const int rr0 = wm * 32 + r + half * 8;
        const int rr1 = rr0 + 16;
        red[rr0 * 64 + lc0]      = gelu_f(acc00[r] + b10) * w20;
        red[rr0 * 64 + lc0 + 16] = gelu_f(acc01[r] + b11) * w21;
        red[rr1 * 64 + lc0]      = gelu_f(acc10[r] + b10) * w20;
        red[rr1 * 64 + lc0 + 16] = gelu_f(acc11[r] + b11) * w21;
    }
    __syncthreads();
    if (tid < 128) {
        const int p = p_blk + tid;
        if (p < NPAIR) {
            float s = 0.f;
            #pragma unroll 8
            for (int c = 0; c < 64; ++c) s += red[tid * 64 + c];
            atomicAdd(&gacc[p], s);
        }
    }
}

// ---------------- final mix: sigmoid gate, combine scores ----------------
__global__ void pair_final_kernel(const float* __restrict__ gacc,
                                  const float* __restrict__ gb2,
                                  const float* __restrict__ ssyn,
                                  const float* __restrict__ sant,
                                  float* __restrict__ out)
{
    int p = blockIdx.x * 256 + threadIdx.x;
    if (p >= NPAIR) return;
    float g = 1.f / (1.f + expf(-(gacc[p] + gb2[0])));
    out[p] = g * sant[p] - (1.f - g) * ssyn[p];
}

// =====================================================================
extern "C" void kernel_launch(void* const* d_in, const int* in_sizes, int n_in,
                              void* d_out, int out_size, void* d_ws, size_t ws_size,
                              hipStream_t stream)
{
    (void)in_sizes; (void)n_in; (void)out_size; (void)ws_size;
    const float* x       = (const float*)d_in[0];
    const int*   eidx    = (const int*)d_in[1];
    const int*   edges   = (const int*)d_in[2];
    const float* mixer_W = (const float*)d_in[3];
    const float* mixer_b = (const float*)d_in[4];
    const float* Wq = (const float*)d_in[5];  const float* bq = (const float*)d_in[6];
    const float* Wk = (const float*)d_in[7];  const float* bk = (const float*)d_in[8];
    const float* Wv = (const float*)d_in[9];  const float* bv = (const float*)d_in[10];
    const float* Ws = (const float*)d_in[11]; const float* bs = (const float*)d_in[12];
    const float* ln_g = (const float*)d_in[13];
    const float* ln_b = (const float*)d_in[14];
    const float* syn_W = (const float*)d_in[15]; const float* syn_b = (const float*)d_in[16];
    const float* ant_W = (const float*)d_in[17]; const float* ant_b = (const float*)d_in[18];
    const float* W_syn = (const float*)d_in[19];
    const float* W_ant = (const float*)d_in[20];
    const float* gW1 = (const float*)d_in[21]; const float* gb1 = (const float*)d_in[22];
    const float* gW2 = (const float*)d_in[23]; const float* gb2 = (const float*)d_in[24];
    float* out = (float*)d_out;

    // ---- workspace carve-up ----
    float* ws = (float*)d_ws;
    size_t off = 0;
    float* h      = ws + off; off += (size_t)NNODE * HID;
    float* q      = ws + off; off += (size_t)NNODE * HID;
    float* k      = ws + off; off += (size_t)NNODE * HID;
    float* v      = ws + off; off += (size_t)NNODE * HID;
    float* skip   = ws + off; off += (size_t)NNODE * HID;   // root term + aggregated msgs
    float* logits = ws + off; off += (size_t)NEDGE * HEADS;
    float* mx     = ws + off; off += (size_t)NNODE * HEADS;
    float* den    = ws + off; off += (size_t)NNODE * HEADS;
    float* z_syn  = ws + off; off += (size_t)NNODE * HID;
    float* z_ant  = ws + off; off += (size_t)NNODE * HID;
    float* T_syn  = ws + off; off += (size_t)NNODE * HID;
    float* T_ant  = ws + off; off += (size_t)NNODE * HID;
    float* s_syn  = ws + off; off += (size_t)NPAIR;
    float* s_ant  = ws + off; off += (size_t)NPAIR;
    float* gacc   = ws + off; off += (size_t)NPAIR;

    const dim3 blk(256);
    const dim3 gemm_grid_h((NNODE + 127) / 128, HID / 64);

    // ---- 1) FeatureMixer: h = gelu(x @ mixer_W + b) ----
    gemm_bias_act_wmma<<<gemm_grid_h, blk, 0, stream>>>(
        x, mixer_W, mixer_b, h, NNODE, HID, INDIM, 1);

    // ---- 2) six TransformerConv layers ----
    const long EH = (long)NEDGE * HEADS;
    const int eh_blocks  = (int)((EH + 255) / 256);
    const int msg_blocks = (int)(((long)NEDGE * (HID / 4) + 255) / 256);
    const int nh_blocks  = (int)(((long)NNODE * HEADS + 255) / 256);

    for (int l = 0; l < LNUM; ++l) {
        const long wl = (long)l * HID * HID;
        const long bl = (long)l * HID;
        gemm_bias_act_wmma<<<gemm_grid_h, blk, 0, stream>>>(h, Wq + wl, bq + bl, q,    NNODE, HID, HID, 0);
        gemm_bias_act_wmma<<<gemm_grid_h, blk, 0, stream>>>(h, Wk + wl, bk + bl, k,    NNODE, HID, HID, 0);
        gemm_bias_act_wmma<<<gemm_grid_h, blk, 0, stream>>>(h, Wv + wl, bv + bl, v,    NNODE, HID, HID, 0);
        gemm_bias_act_wmma<<<gemm_grid_h, blk, 0, stream>>>(h, Ws + wl, bs + bl, skip, NNODE, HID, HID, 0);

        fill_kernel<<<nh_blocks, blk, 0, stream>>>(mx,  (long)NNODE * HEADS, -3.0e38f);
        fill_kernel<<<nh_blocks, blk, 0, stream>>>(den, (long)NNODE * HEADS, 0.f);

        edge_logits_kernel<<<eh_blocks, blk, 0, stream>>>(eidx, q, k, logits, mx);
        edge_exp_kernel   <<<eh_blocks, blk, 0, stream>>>(eidx, logits, mx, den);
        edge_msg_kernel   <<<msg_blocks, blk, 0, stream>>>(eidx, logits, den, v, skip);

        resid_ln_kernel<<<NNODE, blk, 0, stream>>>(skip, h, ln_g + bl, ln_b + bl);
    }

    // ---- 3) dual projections + bilinear pre-transforms ----
    gemm_bias_act_wmma<<<gemm_grid_h, blk, 0, stream>>>(h, syn_W, syn_b, z_syn, NNODE, HID, HID, 1);
    gemm_bias_act_wmma<<<gemm_grid_h, blk, 0, stream>>>(h, ant_W, ant_b, z_ant, NNODE, HID, HID, 1);
    gemm_bias_act_wmma<<<gemm_grid_h, blk, 0, stream>>>(z_syn, W_syn, nullptr, T_syn, NNODE, HID, HID, 0);
    gemm_bias_act_wmma<<<gemm_grid_h, blk, 0, stream>>>(z_ant, W_ant, nullptr, T_ant, NNODE, HID, HID, 0);

    // ---- 4) pair bilinear scores ----
    pair_score_kernel<<<(NPAIR + 255) / 256, blk, 0, stream>>>(
        edges, T_syn, z_syn, T_ant, z_ant, s_syn, s_ant);

    // ---- 5) fused gate MLP (WMMA over on-the-fly pair features) ----
    fill_kernel<<<(NPAIR + 255) / 256, blk, 0, stream>>>(gacc, (long)NPAIR, 0.f);
    dim3 gate_grid((NPAIR + 127) / 128, HID / 64);
    gate_wmma_kernel<<<gate_grid, blk, 0, stream>>>(
        edges, z_syn, z_ant, gW1, gb1, gW2, gacc);

    // ---- 6) final combine ----
    pair_final_kernel<<<(NPAIR + 255) / 256, blk, 0, stream>>>(
        gacc, gb2, s_syn, s_ant, out);
}
